// CrossAttention_22943715295785
// MI455X (gfx1250) — matrix operand order repaired and checked by reading, not compiled, not run on hardware
//
#include <hip/hip_runtime.h>
#include <stdint.h>

#define EMBED 768
#define HEADS 12
#define HDIM  64
#define BQ    4
#define NTOK  4096
#define MTOK  256
#define SCALEQ 0.125f

typedef __attribute__((ext_vector_type(16))) __bf16       v16bf;
typedef __attribute__((ext_vector_type(8)))  float        v8f;
typedef __attribute__((ext_vector_type(4)))  unsigned int u32x4;

union FragU { v16bf bf; u32x4 q[2]; };

__device__ __forceinline__ unsigned short f32_to_bf16(float f) {
  unsigned int u = __float_as_uint(f);
  u += 0x7FFFu + ((u >> 16) & 1u);          // round-to-nearest-even
  return (unsigned short)(u >> 16);
}

// A-matrix 16x32 bf16 fragment (ISA 7.12.2): lane m holds K pairs
// {8h..8h+7} in V0..3 and {16+8h..16+8h+7} in V4..7 (h = lane>=16).
__device__ __forceinline__ v16bf load_a16x32(const unsigned short* row, int k0, int half) {
  FragU f;
  f.q[0] = *(const u32x4*)(row + k0 + 8 * half);
  f.q[1] = *(const u32x4*)(row + k0 + 16 + 8 * half);
  return f.bf;
}
// B-matrix 32x16 bf16 fragment: lane holds its column's K values
// {8h..8h+7} in V0..3 and {16+8h..16+8h+7} in V4..7 (same as A).
__device__ __forceinline__ v16bf load_b32x16(const unsigned short* colrow, int k0, int half) {
  FragU f;
  f.q[0] = *(const u32x4*)(colrow + k0 + 8 * half);
  f.q[1] = *(const u32x4*)(colrow + k0 + 16 + 8 * half);
  return f.bf;
}

#define WMMA_BF16(A, B, C) \
  __builtin_amdgcn_wmma_f32_16x16x32_bf16(false, (A), false, (B), (short)0, (C), false, false)

// ---------------------------------------------------------------- convert
__global__ void cvt_bf16_kernel(const float* __restrict__ in,
                                unsigned short* __restrict__ out, int n) {
  for (int i = blockIdx.x * blockDim.x + threadIdx.x; i < n;
       i += gridDim.x * blockDim.x)
    out[i] = f32_to_bf16(in[i]);
}

// --------------------------------------------------------------- GEMM core
// 16 rows x 96 cols per wave, K=768, software-pipelined: the whole next
// K-step (A + 6 B fragments) is in flight while the current 6 WMMAs issue,
// so s_wait_loadcnt waits leave a K-step of loads outstanding.
__device__ __forceinline__ void gemm16x96(
    const unsigned short* __restrict__ xrow,
    const unsigned short* __restrict__ Wbf,
    int colbase, int ln16, int half, v8f acc[6]) {
  const unsigned short* wrow[6];
#pragma unroll
  for (int t = 0; t < 6; ++t)
    wrow[t] = Wbf + (size_t)(colbase + t * 16 + ln16) * EMBED;

  v16bf A0, A1, B0[6], B1[6];
  A0 = load_a16x32(xrow, 0, half);
#pragma unroll
  for (int t = 0; t < 6; ++t) B0[t] = load_b32x16(wrow[t], 0, half);

  for (int k0 = 0; k0 < EMBED; k0 += 64) {
    A1 = load_a16x32(xrow, k0 + 32, half);
#pragma unroll
    for (int t = 0; t < 6; ++t) B1[t] = load_b32x16(wrow[t], k0 + 32, half);
#pragma unroll
    for (int t = 0; t < 6; ++t) acc[t] = WMMA_BF16(A0, B0[t], acc[t]);
    const int k2 = k0 + 64;
    if (k2 < EMBED) {
      A0 = load_a16x32(xrow, k2, half);
#pragma unroll
      for (int t = 0; t < 6; ++t) B0[t] = load_b32x16(wrow[t], k2, half);
    }
#pragma unroll
    for (int t = 0; t < 6; ++t) acc[t] = WMMA_BF16(A1, B1[t], acc[t]);
  }
}

// ------------------------------------------- Linear + LayerNorm (Q and K)
__global__ __launch_bounds__(256) void proj_ln_kernel(
    const unsigned short* __restrict__ Xbf,   // [rows x 768] bf16
    const unsigned short* __restrict__ Wbf,   // [768 x 768] bf16 (row j = out col j)
    const float* __restrict__ bias,
    const float* __restrict__ gamma,
    const float* __restrict__ beta,
    float outScale,
    unsigned short* __restrict__ Out)         // [rows x 768] bf16
{
  __shared__ float stile[16 * EMBED];
  __shared__ float psum[256];
  __shared__ float psumsq[256];

  const int tid  = threadIdx.x;
  const int lane = tid & 31;
  const int wave = tid >> 5;
  const int half = lane >> 4;
  const int ln16 = lane & 15;
  const int rowbase = blockIdx.x * 16;
  const int colbase = wave * 96;

  v8f acc[6] = {};
  const unsigned short* xrow = Xbf + (size_t)(rowbase + ln16) * EMBED;
  gemm16x96(xrow, Wbf, colbase, ln16, half, acc);

#pragma unroll
  for (int t = 0; t < 6; ++t) {
    int col = colbase + t * 16 + ln16;
    float bv = bias[col];
#pragma unroll
    for (int r = 0; r < 8; ++r)                 // C tile: elem r -> row r+8*half
      stile[(r + 8 * half) * EMBED + col] = acc[t][r] + bv;
  }
  __syncthreads();

  const int rrow = tid >> 4;                    // 0..15 (row in tile)
  const int seg  = tid & 15;                    // 0..15 (48-col segment)
  const float* rp = &stile[rrow * EMBED + seg * 48];
  float s = 0.f, ss = 0.f;
  for (int i = 0; i < 48; ++i) { float v = rp[i]; s += v; ss += v * v; }
  psum[tid] = s; psumsq[tid] = ss;
  __syncthreads();
  float tot = 0.f, tot2 = 0.f;
  for (int i = 0; i < 16; ++i) { tot += psum[rrow * 16 + i]; tot2 += psumsq[rrow * 16 + i]; }
  const float mu  = tot * (1.f / EMBED);
  const float var = tot2 * (1.f / EMBED) - mu * mu;
  const float rs  = __frsqrt_rn(var + 1e-5f);
  unsigned short* orow = Out + (size_t)(rowbase + rrow) * EMBED + seg * 48;
  const float* g  = gamma + seg * 48;
  const float* be = beta  + seg * 48;
  for (int i = 0; i < 48; ++i)
    orow[i] = f32_to_bf16(((rp[i] - mu) * rs * g[i] + be[i]) * outScale);
}

// ------------------------------------------------- Linear (V), transposed
__global__ __launch_bounds__(256) void proj_v_kernel(
    const unsigned short* __restrict__ Xbf,
    const unsigned short* __restrict__ Wbf,
    const float* __restrict__ bias,
    unsigned short* __restrict__ vT)          // [B*768][4096] bf16
{
  const int tid  = threadIdx.x;
  const int lane = tid & 31;
  const int wave = tid >> 5;
  const int half = lane >> 4;
  const int ln16 = lane & 15;
  const int rowbase = blockIdx.x * 16;
  const int colbase = wave * 96;

  v8f acc[6] = {};
  const unsigned short* xrow = Xbf + (size_t)(rowbase + ln16) * EMBED;
  gemm16x96(xrow, Wbf, colbase, ln16, half, acc);

  const int b    = rowbase >> 12;               // token-row / 4096
  const int nloc = (rowbase & 4095) + 8 * half; // first of 8 consecutive n
#pragma unroll
  for (int t = 0; t < 6; ++t) {
    int col = colbase + t * 16 + ln16;
    float bv = bias[col];
    u32x4 st;
#pragma unroll
    for (int r = 0; r < 4; ++r) {
      unsigned int lo = f32_to_bf16(acc[t][2 * r]     + bv);
      unsigned int hi = f32_to_bf16(acc[t][2 * r + 1] + bv);
      st[r] = lo | (hi << 16);
    }
    *(u32x4*)(vT + ((size_t)(b * EMBED + col)) * NTOK + nloc) = st;
  }
}

// ---------------------------------------------------- S = q . k^T logits
// 128 thr = 4 waves; wave: 32 m x 64 n, K=64. All fragments preloaded
// (4 A-tiles + 8 B-tiles), then 16 WMMAs with B reused across both A tiles.
__global__ __launch_bounds__(128) void attn_logits_kernel(
    const unsigned short* __restrict__ Qbf,   // [B*256 x 768] (scaled)
    const unsigned short* __restrict__ Kbf,   // [B*4096 x 768]
    float* __restrict__ S)                    // [B*H, 256, 4096]
{
  const int tid  = threadIdx.x;
  const int lane = tid & 31;
  const int wave = tid >> 5;
  const int half = lane >> 4;
  const int ln16 = lane & 15;
  const int bh = blockIdx.z;
  const int b = bh / HEADS, h = bh % HEADS;
  const int m0 = blockIdx.y * 32;
  const int n0 = blockIdx.x * 256 + wave * 64;

  const unsigned short* qrow[2];
  qrow[0] = Qbf + (size_t)(b * MTOK + m0 + ln16) * EMBED + h * HDIM;
  qrow[1] = qrow[0] + 16 * EMBED;
  const unsigned short* krow[4];
#pragma unroll
  for (int t = 0; t < 4; ++t)
    krow[t] = Kbf + (size_t)(b * NTOK + n0 + t * 16 + ln16) * EMBED + h * HDIM;

  v16bf A[2][2], Bf[4][2];
#pragma unroll
  for (int i = 0; i < 2; ++i)
#pragma unroll
    for (int s = 0; s < 2; ++s) A[i][s] = load_a16x32(qrow[i], s * 32, half);
#pragma unroll
  for (int t = 0; t < 4; ++t)
#pragma unroll
    for (int s = 0; s < 2; ++s) Bf[t][s] = load_b32x16(krow[t], s * 32, half);

  v8f acc[2][4] = {};
#pragma unroll
  for (int s = 0; s < 2; ++s)
#pragma unroll
    for (int i = 0; i < 2; ++i)
#pragma unroll
      for (int t = 0; t < 4; ++t)
        acc[i][t] = WMMA_BF16(A[i][s], Bf[t][s], acc[i][t]);

  float* sbase = S + (size_t)bh * MTOK * NTOK;
#pragma unroll
  for (int i = 0; i < 2; ++i)
#pragma unroll
    for (int t = 0; t < 4; ++t) {
      int n = n0 + t * 16 + ln16;
#pragma unroll
      for (int r = 0; r < 8; ++r) {
        int m = m0 + i * 16 + r + 8 * half;
        sbase[(size_t)m * NTOK + n] = acc[i][t][r];
      }
    }
}

// ----------------------------------------------------- softmax over N=4096
__global__ __launch_bounds__(256) void softmax_kernel(
    float* __restrict__ S, unsigned short* __restrict__ Pbf) {
  __shared__ float red[256];
  const int row = blockIdx.x;
  const int tid = threadIdx.x;
  float* p = S + (size_t)row * NTOK;
  unsigned short* pb = Pbf + (size_t)row * NTOK;

  float v[16];
  float mx = -3.4e38f;
#pragma unroll
  for (int i = 0; i < 16; ++i) { v[i] = p[tid + i * 256]; mx = fmaxf(mx, v[i]); }
  red[tid] = mx; __syncthreads();
  for (int s = 128; s > 0; s >>= 1) {
    if (tid < s) red[tid] = fmaxf(red[tid], red[tid + s]);
    __syncthreads();
  }
  mx = red[0]; __syncthreads();
  float sum = 0.f;
#pragma unroll
  for (int i = 0; i < 16; ++i) { v[i] = __expf(v[i] - mx); sum += v[i]; }
  red[tid] = sum; __syncthreads();
  for (int s = 128; s > 0; s >>= 1) {
    if (tid < s) red[tid] += red[tid + s];
    __syncthreads();
  }
  const float inv = 1.f / red[0];
#pragma unroll
  for (int i = 0; i < 16; ++i) {
    float o = v[i] * inv;
    p[tid + i * 256] = o;
    pb[tid + i * 256] = f32_to_bf16(o);
  }
}

// ------------------------------------- out = P.V + residual (query_tokens)
// 128 thr = 4 waves; wave: 32 m x 64 d, K=4096, double-buffered fragments.
__global__ __launch_bounds__(128) void attn_out_kernel(
    const unsigned short* __restrict__ Pbf,   // [B*H*256 x 4096] bf16
    const unsigned short* __restrict__ vT,    // [B*768 x 4096] bf16
    const float* __restrict__ qtok,           // [B,256,768]
    float* __restrict__ Out)                  // [B,256,768]
{
  const int tid  = threadIdx.x;
  const int lane = tid & 31;
  const int wave = tid >> 5;
  const int half = lane >> 4;
  const int ln16 = lane & 15;
  const int bh = blockIdx.y;
  const int b = bh / HEADS, h = bh % HEADS;
  const int m0 = blockIdx.x * 128 + wave * 32;

  const unsigned short* prow[2];
  prow[0] = Pbf + ((size_t)bh * MTOK + m0 + ln16) * NTOK;
  prow[1] = prow[0] + (size_t)16 * NTOK;
  const unsigned short* vrow[4];
#pragma unroll
  for (int t = 0; t < 4; ++t)
    vrow[t] = vT + (size_t)(b * EMBED + h * HDIM + t * 16 + ln16) * NTOK;

  v8f acc[2][4] = {};
  v16bf A0[2], A1[2], B0[4], B1[4];
#pragma unroll
  for (int i = 0; i < 2; ++i) A0[i] = load_a16x32(prow[i], 0, half);
#pragma unroll
  for (int t = 0; t < 4; ++t) B0[t] = load_b32x16(vrow[t], 0, half);

  for (int k0 = 0; k0 < NTOK; k0 += 64) {
    __builtin_prefetch(prow[0] + k0 + 1024, 0, 0);   // global_prefetch_b8
#pragma unroll
    for (int i = 0; i < 2; ++i) A1[i] = load_a16x32(prow[i], k0 + 32, half);
#pragma unroll
    for (int t = 0; t < 4; ++t) B1[t] = load_b32x16(vrow[t], k0 + 32, half);
#pragma unroll
    for (int i = 0; i < 2; ++i)
#pragma unroll
      for (int t = 0; t < 4; ++t) acc[i][t] = WMMA_BF16(A0[i], B0[t], acc[i][t]);
    const int k2 = k0 + 64;
    if (k2 < NTOK) {
#pragma unroll
      for (int i = 0; i < 2; ++i) A0[i] = load_a16x32(prow[i], k2, half);
#pragma unroll
      for (int t = 0; t < 4; ++t) B0[t] = load_b32x16(vrow[t], k2, half);
    }
#pragma unroll
    for (int i = 0; i < 2; ++i)
#pragma unroll
      for (int t = 0; t < 4; ++t) acc[i][t] = WMMA_BF16(A1[i], B1[t], acc[i][t]);
  }

#pragma unroll
  for (int i = 0; i < 2; ++i)
#pragma unroll
    for (int t = 0; t < 4; ++t) {
      int c = h * HDIM + t * 16 + ln16;
#pragma unroll
      for (int r = 0; r < 8; ++r) {
        int m = m0 + i * 16 + r + 8 * half;
        size_t idx = ((size_t)b * MTOK + m) * EMBED + c;
        Out[idx] = qtok[idx] + acc[i][t][r];
      }
    }
}

// ------------------------------------------------------------------ launch
extern "C" void kernel_launch(void* const* d_in, const int* in_sizes, int n_in,
                              void* d_out, int out_size, void* d_ws, size_t ws_size,
                              hipStream_t stream) {
  const float* x    = (const float*)d_in[0];   // input_tokens [4,4096,768]
  const float* qt   = (const float*)d_in[1];   // query_tokens [4,256,768]
  const float* Wq   = (const float*)d_in[2];
  const float* bq   = (const float*)d_in[3];
  const float* Wk   = (const float*)d_in[4];
  const float* bk   = (const float*)d_in[5];
  const float* Wv   = (const float*)d_in[6];
  const float* bv   = (const float*)d_in[7];
  const float* gq   = (const float*)d_in[8];
  const float* bqln = (const float*)d_in[9];
  const float* gk   = (const float*)d_in[10];
  const float* bkln = (const float*)d_in[11];

  float* out  = (float*)d_out;                       // [B,M,C]
  float* attn = out + (size_t)BQ * MTOK * EMBED;     // [B,H,M,N]

  unsigned short* ws = (unsigned short*)d_ws;
  size_t o = 0;
  unsigned short* wqbf = ws + o; o += (size_t)EMBED * EMBED;
  unsigned short* wkbf = ws + o; o += (size_t)EMBED * EMBED;
  unsigned short* wvbf = ws + o; o += (size_t)EMBED * EMBED;
  unsigned short* xbf  = ws + o; o += (size_t)BQ * NTOK * EMBED;
  unsigned short* qtbf = ws + o; o += (size_t)BQ * MTOK * EMBED;
  unsigned short* qbf  = ws + o; o += (size_t)BQ * MTOK * EMBED;
  unsigned short* kbf  = ws + o; o += (size_t)BQ * NTOK * EMBED;
  unsigned short* vTb  = ws + o; o += (size_t)BQ * EMBED * NTOK;
  unsigned short* pbf  = ws + o; o += (size_t)BQ * HEADS * MTOK * NTOK;

  auto cvt = [&](const float* src, unsigned short* dst, int n) {
    int blocks = (n + 255) / 256;
    if (blocks > 16384) blocks = 16384;
    cvt_bf16_kernel<<<blocks, 256, 0, stream>>>(src, dst, n);
  };
  cvt(Wq, wqbf, EMBED * EMBED);
  cvt(Wk, wkbf, EMBED * EMBED);
  cvt(Wv, wvbf, EMBED * EMBED);
  cvt(x,  xbf,  BQ * NTOK * EMBED);
  cvt(qt, qtbf, BQ * MTOK * EMBED);

  proj_ln_kernel<<<(BQ * MTOK) / 16, 256, 0, stream>>>(qtbf, wqbf, bq, gq, bqln, SCALEQ, qbf);
  proj_ln_kernel<<<(BQ * NTOK) / 16, 256, 0, stream>>>(xbf, wkbf, bk, gk, bkln, 1.0f, kbf);
  proj_v_kernel <<<(BQ * NTOK) / 16, 256, 0, stream>>>(xbf, wvbf, bv, vTb);

  attn_logits_kernel<<<dim3(NTOK / 256, MTOK / 32, BQ * HEADS), 128, 0, stream>>>(qbf, kbf, attn);
  softmax_kernel<<<BQ * HEADS * MTOK, 256, 0, stream>>>(attn, pbf);
  attn_out_kernel<<<dim3(MTOK / 128, BQ * HEADS), 128, 0, stream>>>(pbf, vTb, qt, out);
}
